// GptOssGroupedExperts_12309376271017
// MI455X (gfx1250) — matrix operand order repaired and checked by reading, not compiled, not run on hardware
//
#include <hip/hip_runtime.h>
#include <cstdint>
#include <cstddef>

// ---------------------------------------------------------------------------
// MoE grouped experts (gpt-oss style) for MI455X / gfx1250.
// bf16 WMMA GEMMs, native bf16 converts, register-staged software pipeline,
// async global->LDS copies for the bf16 activation tile in GEMM2.
// ---------------------------------------------------------------------------

typedef __attribute__((ext_vector_type(16))) __bf16       v16bf;
typedef __attribute__((ext_vector_type(2)))  __bf16       v2bf;
typedef __attribute__((ext_vector_type(8)))  float        v8f;
typedef __attribute__((ext_vector_type(4)))  float        v4f;
typedef __attribute__((ext_vector_type(4)))  unsigned int v4u;
typedef __attribute__((ext_vector_type(2)))  unsigned int v2u;
typedef __attribute__((ext_vector_type(4)))  int          v4i;

#ifndef __has_builtin
#define __has_builtin(x) 0
#endif

#if __has_builtin(__builtin_amdgcn_global_load_async_to_lds_b128)
#define HAVE_ASYNC_LDS 1
#else
#define HAVE_ASYNC_LDS 0
#endif

#define GLOBAL_AS __attribute__((address_space(1)))
#define LDS_AS    __attribute__((address_space(3)))

#define NEXP 16
#define TPE  2048
#define DIM  2880
#define HID  2880
#define TWOH 5760

#define BM 128
#define BN 96
#define BK 64
#define LDS_STRIDE 36   // uints per row (32 data + 4 pad) -> bank-conflict-free b128

// Native fp32 -> packed bf16 (RNE). Prefer v_cvt_pk_bf16_f32.
__device__ __forceinline__ unsigned int pack2(float lo, float hi) {
#if __has_builtin(__builtin_amdgcn_cvt_pk_bf16_f32)
  union { v2bf h; unsigned int u; } c;
  c.h = __builtin_amdgcn_cvt_pk_bf16_f32(lo, hi);
  return c.u;
#else
  union { v2bf h; unsigned int u; } c;
  c.h[0] = (__bf16)lo; c.h[1] = (__bf16)hi;
  return c.u;
#endif
}
__device__ __forceinline__ unsigned short f2bf16(float f) {
  union { __bf16 h; unsigned short s; } c;
  c.h = (__bf16)f;
  return c.s;
}

#if HAVE_ASYNC_LDS
__device__ __forceinline__ void async_copy_b128(const void* g, void* l) {
  __builtin_amdgcn_global_load_async_to_lds_b128(
      (v4i*)g, (LDS_AS v4i*)l, /*offset=*/0, /*cpol=*/0);
}
#endif

__device__ __forceinline__ void wait_async0() {
#if HAVE_ASYNC_LDS
#if __has_builtin(__builtin_amdgcn_s_wait_asynccnt)
  __builtin_amdgcn_s_wait_asynccnt(0);
#else
  asm volatile("s_wait_asynccnt 0" ::: "memory");
#endif
#endif
}

union Frag { v16bf v; v4u q[2]; };

// ============================ GEMM1 + SwiGLU ===============================
// h = x @ W1^T + b1 ; hact = swiglu(h) stored bf16 [T, HID]
// W1 is [E, 2H, D] row-major: row 2n -> glu column n, row 2n+1 -> lin column n.
__global__ __launch_bounds__(256)
void moe_mlp1_swiglu(const float* __restrict__ x,
                     const float* __restrict__ w1,
                     const float* __restrict__ b1,
                     unsigned short* __restrict__ hact)
{
  __shared__ uint32_t sA [BM * LDS_STRIDE];
  __shared__ uint32_t sBg[BN * LDS_STRIDE];
  __shared__ uint32_t sBl[BN * LDS_STRIDE];

  const int e    = blockIdx.z;
  const int tok0 = e * TPE + blockIdx.y * BM;
  const int n0   = blockIdx.x * BN;

  const int tid  = threadIdx.x;
  const int lane = tid & 31;
  const int lr   = lane & 15;
  const int kh   = lane >> 4;
  const int wm   = (tid >> 5) >> 1;   // 0..3 : 32 rows each
  const int wn   = (tid >> 5) & 1;    // 0..1 : 48 cols each
  const int arow = tid >> 4;          // 0..15 staging row
  const int ac4  = tid & 15;          // staging float4 column

  v8f accg[2][3], accl[2][3];
  const v8f vzero = {0.f,0.f,0.f,0.f,0.f,0.f,0.f,0.f};
#pragma unroll
  for (int i = 0; i < 2; ++i)
#pragma unroll
    for (int j = 0; j < 3; ++j) { accg[i][j] = vzero; accl[i][j] = vzero; }

  const float* xb  = x  + (size_t)tok0 * DIM;
  const float* w1b = w1 + (size_t)e * TWOH * DIM;

  // -------- software pipeline: stage K-tile 0 into registers --------
  v4f pa[8], pg[6], pl[6];
#pragma unroll
  for (int p = 0; p < 8; ++p)
    pa[p] = *(const v4f*)(xb + (size_t)(p * 16 + arow) * DIM + ac4 * 4);
#pragma unroll
  for (int p = 0; p < 6; ++p) {
    const float* rg = w1b + (size_t)(2 * (n0 + p * 16 + arow)) * DIM + ac4 * 4;
    pg[p] = *(const v4f*)rg;
    pl[p] = *(const v4f*)(rg + DIM);
  }

  for (int kt = 0; kt < DIM; kt += BK) {
    // ---- commit staged tile to LDS as bf16 ----
#pragma unroll
    for (int p = 0; p < 8; ++p) {
      v2u pk; pk.x = pack2(pa[p].x, pa[p].y); pk.y = pack2(pa[p].z, pa[p].w);
      *(v2u*)&sA[(p * 16 + arow) * LDS_STRIDE + ac4 * 2] = pk;
    }
#pragma unroll
    for (int p = 0; p < 6; ++p) {
      v2u g; g.x = pack2(pg[p].x, pg[p].y); g.y = pack2(pg[p].z, pg[p].w);
      v2u l; l.x = pack2(pl[p].x, pl[p].y); l.y = pack2(pl[p].z, pl[p].w);
      *(v2u*)&sBg[(p * 16 + arow) * LDS_STRIDE + ac4 * 2] = g;
      *(v2u*)&sBl[(p * 16 + arow) * LDS_STRIDE + ac4 * 2] = l;
    }
    // ---- issue next tile's global loads; they complete under the WMMAs ----
    const int kn = kt + BK;
    if (kn < DIM) {
#pragma unroll
      for (int p = 0; p < 8; ++p)
        pa[p] = *(const v4f*)(xb + (size_t)(p * 16 + arow) * DIM + kn + ac4 * 4);
#pragma unroll
      for (int p = 0; p < 6; ++p) {
        const float* rg = w1b + (size_t)(2 * (n0 + p * 16 + arow)) * DIM + kn + ac4 * 4;
        pg[p] = *(const v4f*)rg;
        pl[p] = *(const v4f*)(rg + DIM);
      }
    }
    __syncthreads();

    // ---- compute: two K=32 WMMA steps from LDS ----
#pragma unroll
    for (int kk = 0; kk < 2; ++kk) {
      Frag a[2];
#pragma unroll
      for (int i = 0; i < 2; ++i) {
        const int row = wm * 32 + i * 16 + lr;
        const uint32_t* base = &sA[row * LDS_STRIDE + kk * 16 + kh * 4];
        a[i].q[0] = *(const v4u*)base;        // K = kh*8 .. +7
        a[i].q[1] = *(const v4u*)(base + 8);  // K = 16+kh*8 .. +7
      }
      Frag bg[3], bl[3];
#pragma unroll
      for (int j = 0; j < 3; ++j) {
        const int col = wn * 48 + j * 16 + lr;
        const uint32_t* pgp = &sBg[col * LDS_STRIDE + kk * 16 + kh * 4];
        bg[j].q[0] = *(const v4u*)pgp;        // K = kh*8 .. +7, then 16+kh*8 .. +7
        bg[j].q[1] = *(const v4u*)(pgp + 8);
        const uint32_t* plp = &sBl[col * LDS_STRIDE + kk * 16 + kh * 4];
        bl[j].q[0] = *(const v4u*)plp;
        bl[j].q[1] = *(const v4u*)(plp + 8);
      }
#pragma unroll
      for (int i = 0; i < 2; ++i)
#pragma unroll
        for (int j = 0; j < 3; ++j) {
          accg[i][j] = __builtin_amdgcn_wmma_f32_16x16x32_bf16(
              false, a[i].v, false, bg[j].v, (short)0, accg[i][j], false, false);
          accl[i][j] = __builtin_amdgcn_wmma_f32_16x16x32_bf16(
              false, a[i].v, false, bl[j].v, (short)0, accl[i][j], false, false);
        }
    }
    __syncthreads();
  }

  // ---- epilogue: bias + SwiGLU, store bf16 activations ----
  const float* b1e = b1 + (size_t)e * TWOH;
#pragma unroll
  for (int j = 0; j < 3; ++j) {
    const int col = n0 + wn * 48 + j * 16 + lr;
    const float bgv = b1e[2 * col];
    const float blv = b1e[2 * col + 1];
#pragma unroll
    for (int i = 0; i < 2; ++i) {
      const int rowb = tok0 + wm * 32 + i * 16 + kh * 8;
#pragma unroll
      for (int r = 0; r < 8; ++r) {
        float g = accg[i][j][r] + bgv;
        float l = accl[i][j][r] + blv;
        g = fminf(g, 7.0f);                       // clamp max only
        l = fminf(fmaxf(l, -7.0f), 7.0f);         // clamp both
        const float s = 1.0f / (1.0f + __expf(-1.702f * g));
        const float act = g * s * (l + 1.0f);
        hact[(size_t)(rowb + r) * HID + col] = f2bf16(act);
      }
    }
  }
}

// ================================ GEMM2 ====================================
// out = hact @ W2^T + b2 ; W2 is [E, D, H] row-major (row n = output col n).
__global__ __launch_bounds__(256)
void moe_mlp2(const unsigned short* __restrict__ hact,
              const float* __restrict__ w2,
              const float* __restrict__ b2,
              float* __restrict__ out)
{
  __shared__ uint32_t sA[BM * LDS_STRIDE];
  __shared__ uint32_t sB[BN * LDS_STRIDE];

  const int e    = blockIdx.z;
  const int tok0 = e * TPE + blockIdx.y * BM;
  const int n0   = blockIdx.x * BN;

  const int tid  = threadIdx.x;
  const int lane = tid & 31;
  const int lr   = lane & 15;
  const int kh   = lane >> 4;
  const int wm   = (tid >> 5) >> 1;
  const int wn   = (tid >> 5) & 1;
  const int arow = tid >> 4;
  const int ac4  = tid & 15;

  v8f acc[2][3];
  const v8f vzero = {0.f,0.f,0.f,0.f,0.f,0.f,0.f,0.f};
#pragma unroll
  for (int i = 0; i < 2; ++i)
#pragma unroll
    for (int j = 0; j < 3; ++j) acc[i][j] = vzero;

  const unsigned short* hb = hact + (size_t)tok0 * HID;
  const float* w2b = w2 + (size_t)e * DIM * HID;

  // -------- pipeline prologue: stage B K-tile 0 (fp32) --------
  v4f pb[6];
#pragma unroll
  for (int p = 0; p < 6; ++p)
    pb[p] = *(const v4f*)(w2b + (size_t)(n0 + p * 16 + arow) * HID + ac4 * 4);

  for (int kt = 0; kt < HID; kt += BK) {
    // ---- A tile: bf16 copy global->LDS (async path when available) ----
#if HAVE_ASYNC_LDS
#pragma unroll
    for (int p = 0; p < 4; ++p) {
      const int row = p * 32 + (tid >> 3);
      const int c4  = tid & 7;
      async_copy_b128(hb + (size_t)row * HID + kt + c4 * 8,
                      &sA[row * LDS_STRIDE + c4 * 4]);
    }
#else
#pragma unroll
    for (int p = 0; p < 4; ++p) {
      const int row = p * 32 + (tid >> 3);
      const int c4  = tid & 7;
      v4u v = *(const v4u*)(hb + (size_t)row * HID + kt + c4 * 8);
      *(v4u*)&sA[row * LDS_STRIDE + c4 * 4] = v;
    }
#endif
    // ---- commit staged B tile to LDS as bf16 ----
#pragma unroll
    for (int p = 0; p < 6; ++p) {
      v2u pk; pk.x = pack2(pb[p].x, pb[p].y); pk.y = pack2(pb[p].z, pb[p].w);
      *(v2u*)&sB[(p * 16 + arow) * LDS_STRIDE + ac4 * 2] = pk;
    }
    // ---- prefetch next B tile ----
    const int kn = kt + BK;
    if (kn < HID) {
#pragma unroll
      for (int p = 0; p < 6; ++p)
        pb[p] = *(const v4f*)(w2b + (size_t)(n0 + p * 16 + arow) * HID + kn + ac4 * 4);
    }
    wait_async0();          // my async A chunks are in LDS before I signal
    __syncthreads();

#pragma unroll
    for (int kk = 0; kk < 2; ++kk) {
      Frag a[2];
#pragma unroll
      for (int i = 0; i < 2; ++i) {
        const int row = wm * 32 + i * 16 + lr;
        const uint32_t* base = &sA[row * LDS_STRIDE + kk * 16 + kh * 4];
        a[i].q[0] = *(const v4u*)base;
        a[i].q[1] = *(const v4u*)(base + 8);
      }
      Frag bfrag[3];
#pragma unroll
      for (int j = 0; j < 3; ++j) {
        const int col = wn * 48 + j * 16 + lr;
        const uint32_t* pbp = &sB[col * LDS_STRIDE + kk * 16 + kh * 4];
        bfrag[j].q[0] = *(const v4u*)pbp;
        bfrag[j].q[1] = *(const v4u*)(pbp + 8);
      }
#pragma unroll
      for (int i = 0; i < 2; ++i)
#pragma unroll
        for (int j = 0; j < 3; ++j)
          acc[i][j] = __builtin_amdgcn_wmma_f32_16x16x32_bf16(
              false, a[i].v, false, bfrag[j].v, (short)0, acc[i][j], false, false);
    }
    __syncthreads();
  }

  const float* b2e = b2 + (size_t)e * DIM;
#pragma unroll
  for (int j = 0; j < 3; ++j) {
    const int col = n0 + wn * 48 + j * 16 + lr;
    const float bv = b2e[col];
#pragma unroll
    for (int i = 0; i < 2; ++i) {
      const int rowb = tok0 + wm * 32 + i * 16 + kh * 8;
#pragma unroll
      for (int r = 0; r < 8; ++r)
        out[(size_t)(rowb + r) * DIM + col] = acc[i][j][r] + bv;
    }
  }
}

// ================================ launch ===================================
extern "C" void kernel_launch(void* const* d_in, const int* in_sizes, int n_in,
                              void* d_out, int out_size, void* d_ws, size_t ws_size,
                              hipStream_t stream) {
  const float* x   = (const float*)d_in[0];
  const float* w1  = (const float*)d_in[1];
  const float* b1  = (const float*)d_in[2];
  const float* w2  = (const float*)d_in[3];
  const float* b2  = (const float*)d_in[4];
  // d_in[5] = num_tokens_per_expert: uniform TPE=2048 by construction.

  unsigned short* hact = (unsigned short*)d_ws;   // [T, HID] bf16 (189 MB)

  dim3 blk(256);
  dim3 g1(HID / BN, TPE / BM, NEXP);   // (30, 16, 16)
  moe_mlp1_swiglu<<<g1, blk, 0, stream>>>(x, w1, b1, hact);

  dim3 g2(DIM / BN, TPE / BM, NEXP);   // (30, 16, 16)
  moe_mlp2<<<g2, blk, 0, stream>>>(hact, w2, b2, (float*)d_out);
}
